// RaindropSimulator_40338332844373
// MI455X (gfx1250) — matrix operand order repaired and checked by reading, not compiled, not run on hardware
//
#include <hip/hip_runtime.h>
#include <hip/hip_bf16.h>
#include <math.h>
#include <stdint.h>

typedef __attribute__((ext_vector_type(16))) _Float16 v16h;
typedef __attribute__((ext_vector_type(8)))  float    v8f;

#define HDIM 1024
#define WDIM 1024
#define HW   (1024*1024)

// padded `enhanced` plane: 7-element zero guard band on all sides, stride 1040
#define ENH_STRIDE 1040
#define ENH_ROWS   1040
#define ENH_PLANE  (ENH_STRIDE * ENH_ROWS)
#define ENH_PAD    7

// ---------------------------------------------------------------------------
// Kernel 1: scatter dot indices, last-write-wins via atomicMax on dot index.
// ---------------------------------------------------------------------------
__global__ __launch_bounds__(256) void scatter_dots_k(
    const int* __restrict__ dot_y, const int* __restrict__ dot_x,
    int nd, int total, int* __restrict__ dotmap)
{
    int j = blockIdx.x * 256 + threadIdx.x;
    if (j >= total) return;
    int n = j / nd;
    int i = j - n * nd;
    int y = dot_y[j];
    int x = dot_x[j];
    atomicMax(&dotmap[(n * HDIM + y) * WDIM + x], i);
}

// ---------------------------------------------------------------------------
// Kernel 2: fused stripe pattern + dot override + sigmoid -> rain_drops (f32)
// ---------------------------------------------------------------------------
__global__ __launch_bounds__(256) void rain_pattern_k(
    const float* __restrict__ params, const float* __restrict__ angle,
    const float* __restrict__ stripe_rand, const float* __restrict__ rvar,
    const float* __restrict__ dot_val, const int* __restrict__ dotmap,
    int nd, float* __restrict__ rain)
{
    int idx = blockIdx.x * 256 + threadIdx.x;      // [0, 4*HW)
    int n = idx >> 20;
    int y = (idx >> 10) & 1023;
    int x = idx & 1023;

    float ang = angle[n] * 0.1f - 0.05f;
    float sa = sinf(ang), ca = cosf(ang);
    float stripe_pos = ((float)x * sa + (float)y * ca) * 0.08f;
    float base = (sinf(stripe_pos) > 0.9f) ? stripe_rand[idx] * 4.0f : 0.0f;

    int di = dotmap[idx];
    if (di >= 0) base = dot_val[n * nd + di] * 5.0f;

    float rp = base + rvar[idx] * 0.5f;
    float density = 1.0f / (1.0f + expf(-0.5f * params[n * 4 + 0])) * 0.8f + 0.2f;
    rain[idx] = density / (1.0f + expf(-8.0f * rp));
}

// ---------------------------------------------------------------------------
// Kernel 3: fused texture enhancer: conv3x3(1->16)+ReLU, conv1x1(16->1)+sigmoid
// Writes f16 into the zero-padded `enhanced` plane (guard band pre-zeroed).
// ---------------------------------------------------------------------------
__global__ __launch_bounds__(256) void enhancer_k(
    const float* __restrict__ rain,
    const float* __restrict__ w1, const float* __restrict__ b1,
    const float* __restrict__ w2, const float* __restrict__ b2,
    _Float16* __restrict__ enh)
{
    __shared__ float tile[18 * 18];
    __shared__ float wl[144];
    __shared__ float b1l[16], w2l[16];
    __shared__ float b2l;

    int tx = threadIdx.x, ty = threadIdx.y;
    int tid = ty * 16 + tx;
    int n  = blockIdx.z;
    int x0 = blockIdx.x * 16;
    int y0 = blockIdx.y * 16;
    const float* rn = rain + (size_t)n * HW;

    for (int idx = tid; idx < 18 * 18; idx += 256) {
        int r  = idx / 18;
        int cc = idx - r * 18;
        int gy = y0 + r - 1, gx = x0 + cc - 1;
        tile[idx] = (gy >= 0 && gy < HDIM && gx >= 0 && gx < WDIM)
                        ? rn[gy * WDIM + gx] : 0.0f;
    }
    if (tid < 144) wl[tid] = w1[tid];
    if (tid < 16) { b1l[tid] = b1[tid]; w2l[tid] = w2[tid]; }
    if (tid == 0) b2l = b2[0];
    __syncthreads();

    float v[9];
#pragma unroll
    for (int t = 0; t < 9; ++t) {
        int ky = t / 3, kx = t - 3 * ky;
        v[t] = tile[(ty + ky) * 18 + (tx + kx)];
    }
    float acc = b2l;
#pragma unroll
    for (int ch = 0; ch < 16; ++ch) {
        float s = b1l[ch];
#pragma unroll
        for (int t = 0; t < 9; ++t) s = fmaf(wl[ch * 9 + t], v[t], s);
        acc = fmaf(fmaxf(s, 0.0f), w2l[ch], acc);
    }
    float e = 1.0f / (1.0f + expf(-acc));
    enh[(size_t)n * ENH_PLANE + (size_t)(y0 + ty + ENH_PAD) * ENH_STRIDE
        + (x0 + tx + ENH_PAD)] = (_Float16)e;
}

// ---------------------------------------------------------------------------
// Kernel 4: build the 4 motion kernels (exact overwrite/total semantics) and
// pack as the WMMA B matrix: per 32-K chunk, lanes 0-15 hold K=0..15 and
// lanes 16-31 hold K=16..31, K ascending across the 16 halves (contiguous K order, same as A).
// ---------------------------------------------------------------------------
__global__ void motion_pack_k(const float* __restrict__ params,
                              _Float16* __restrict__ Bbuf)
{
    __shared__ float kern[4][225];
    __shared__ float tot[4];
    int lane = threadIdx.x;   // 32 threads

    for (int i = lane; i < 4 * 225; i += 32) (&kern[0][0])[i] = 0.0f;
    __syncthreads();

    if (lane < 4) {
        int o = lane;
        float dxv  = tanhf(0.5f * params[o * 4 + 2]) * 6.0f;
        float dyv  = tanhf(0.5f * params[o * 4 + 3]) * 6.0f;
        float drop = 1.0f / (1.0f + expf(-0.5f * params[o * 4 + 1])) * 3.0f + 2.0f;
        float lenf = floorf(fminf(6.0f, drop * 2.0f));
        int length = (int)lenf;
        float total = 0.0f;
        for (int t = -6; t <= 6; ++t) {
            int at = (t < 0) ? -t : t;
            float wgt = (t == 0) ? 1.0f : expf((float)(-at) * 1.5f / lenf);
            int xp = 7 + (int)((float)t * dxv);   // trunc toward zero
            int yp = 7 + (int)((float)t * dyv);
            bool valid = (at <= length) && xp >= 0 && xp < 15 && yp >= 0 && yp < 15;
            if (valid) { kern[o][yp * 15 + xp] = wgt; total += wgt; }
        }
        tot[o] = total;
    }
    __syncthreads();

    for (int i = lane; i < 4 * 225; i += 32) {
        int o = i / 225, k = i - o * 225;
        float t = tot[o];
        float val = (t > 0.0f) ? kern[o][k] / t : ((k == 7 * 15 + 7) ? 1.0f : 0.0f);
        kern[o][k] = val;
    }
    __syncthreads();

    int koff = (lane < 16) ? 0 : 16;
    int o = lane & 15;
    for (int c = 0; c < 8; ++c) {
        for (int h = 0; h < 16; ++h) {
            int kt = c * 32 + koff + h;
            float val = (o < 4 && kt < 225) ? kern[o][kt] : 0.0f;
            Bbuf[c * 512 + lane * 16 + h] = (_Float16)val;
        }
    }
}

// ---------------------------------------------------------------------------
// Kernel 5: 15x15 cross-batch motion conv via v_wmma_f32_16x16x32_f16.
// Halo tile (22x30 f16) is staged with CDNA5 async global->LDS loads
// (global_load_async_to_lds_b32, ASYNCcnt, s_wait_asynccnt) from the
// zero-padded enhanced plane, so no bounds checks / no VGPR round-trips.
// Epilogue transposes D through LDS for 64B-contiguous coalesced stores.
// ---------------------------------------------------------------------------
__global__ __launch_bounds__(256) void conv_wmma_k(
    const _Float16* __restrict__ enh, const _Float16* __restrict__ Bbuf,
    const float* __restrict__ intensity, float* __restrict__ out)
{
    __shared__ _Float16 ebuf[22 * 32];
    __shared__ float sbuf[8][4][16];

    int n  = blockIdx.z;
    int x0 = blockIdx.x * 16;
    int y0 = blockIdx.y * 8;
    int tid = threadIdx.x;

    // ---- async fill: 22 rows x 15 b32 (pairs of halves) = 330 transfers ----
    const _Float16* gb = enh + (size_t)n * ENH_PLANE;   // uniform SGPR base
    uint32_t ldsbase = (uint32_t)(uintptr_t)(&ebuf[0]);
    for (int idx = tid; idx < 22 * 15; idx += 256) {
        int r = idx / 15;
        int c = idx - r * 15;
        uint32_t ldsaddr = ldsbase + (uint32_t)((r * 32 + 2 * c) * 2);
        uint32_t voff = (uint32_t)(((y0 + r) * ENH_STRIDE + (x0 + 2 * c)) * 2);
        asm volatile("global_load_async_to_lds_b32 %0, %1, %2"
                     :: "v"(ldsaddr), "v"(voff), "s"(gb)
                     : "memory");
    }
    asm volatile("s_wait_asynccnt 0x0" ::: "memory");
    __syncthreads();

    int lane = tid & 31;
    int w    = tid >> 5;                  // wave id == row offset in block
    int m    = lane & 15;                 // A-matrix row M
    int kadd = (lane < 16) ? 0 : 16;      // A-layout K offset per half-wave

    v8f acc = {};
#pragma unroll
    for (int c = 0; c < 8; ++c) {
        v16h a;
#pragma unroll
        for (int h = 0; h < 16; ++h) {
            int kk = h + kadd;                       // K within chunk
            int kt = c * 32 + kk;                    // global tap index
            int ky = kt / 15;
            int kx = kt - ky * 15;
            a[h] = (kt < 225) ? ebuf[(w + ky) * 32 + (m + kx)] : (_Float16)0.f;
        }
        v16h bm = *(const v16h*)(Bbuf + c * 512 + lane * 16);
        acc = __builtin_amdgcn_wmma_f32_16x16x32_f16(
            false, a, false, bm, (short)0, acc, false, false);
    }

    // ---- D layout: col N = lane&15, row M = r + 8*(lane>=16) -> LDS ----
    int o = lane & 15;
    if (o < 4) {
#pragma unroll
        for (int r = 0; r < 8; ++r) {
            int mm = r + ((lane >= 16) ? 8 : 0);
            sbuf[w][o][mm] = acc[r];
        }
    }
    __syncthreads();

    // ---- coalesced stores: 512 pixels x 3 channel-tile copies ----
    float scale = intensity[n] * 10.0f;   // (*2 intensity) * 5
#pragma unroll
    for (int pass = 0; pass < 2; ++pass) {
        int idx = pass * 256 + tid;       // [0, 512)
        int xl = idx & 15;
        int oo = (idx >> 4) & 3;
        int wy = idx >> 6;
        float v = fminf(fmaxf(sbuf[wy][oo][xl] * scale, 0.0f), 2.0f);
        size_t base = (((size_t)n * 12 + oo) * HDIM + (y0 + wy)) * WDIM + (x0 + xl);
        out[base]                   = v;
        out[base + (size_t)4 * HW]  = v;
        out[base + (size_t)8 * HW]  = v;
    }
}

// ---------------------------------------------------------------------------
// Launch. Inputs (setup_inputs order):
// 0:x(unused) 1:intensity 2:params 3:random_variation 4:angle 5:stripe_rand
// 6:dot_val 7:w1 8:b1 9:w2 10:b2 11:dot_y 12:dot_x
// Workspace: [0,16Mi) dotmap i32 | [16Mi,32Mi) rain f32 |
//            [32Mi, +8.26Mi) padded enhanced f16 | [41Mi,+8Ki) packed B f16
// ---------------------------------------------------------------------------
extern "C" void kernel_launch(void* const* d_in, const int* in_sizes, int n_in,
                              void* d_out, int out_size, void* d_ws, size_t ws_size,
                              hipStream_t stream) {
    const float* intensity = (const float*)d_in[1];
    const float* params    = (const float*)d_in[2];
    const float* rvar      = (const float*)d_in[3];
    const float* angle     = (const float*)d_in[4];
    const float* stripe    = (const float*)d_in[5];
    const float* dot_val   = (const float*)d_in[6];
    const float* w1        = (const float*)d_in[7];
    const float* b1        = (const float*)d_in[8];
    const float* w2        = (const float*)d_in[9];
    const float* b2        = (const float*)d_in[10];
    const int*   dot_y     = (const int*)d_in[11];
    const int*   dot_x     = (const int*)d_in[12];

    int B  = in_sizes[1];                 // 4
    int nd = in_sizes[6] / B;             // 31457

    char* ws = (char*)d_ws;
    int*      dotmap = (int*)ws;
    float*    rain   = (float*)(ws + (size_t)16 * 1024 * 1024);
    _Float16* enh    = (_Float16*)(ws + (size_t)32 * 1024 * 1024);
    _Float16* Bbuf   = (_Float16*)(ws + (size_t)41 * 1024 * 1024);

    hipMemsetAsync(dotmap, 0xFF, (size_t)B * HW * sizeof(int), stream);
    hipMemsetAsync(enh, 0x00, (size_t)B * ENH_PLANE * sizeof(_Float16), stream);

    int total = B * nd;
    scatter_dots_k<<<(total + 255) / 256, 256, 0, stream>>>(dot_y, dot_x, nd, total, dotmap);

    rain_pattern_k<<<(B * HW) / 256, 256, 0, stream>>>(
        params, angle, stripe, rvar, dot_val, dotmap, nd, rain);

    enhancer_k<<<dim3(WDIM / 16, HDIM / 16, B), dim3(16, 16), 0, stream>>>(
        rain, w1, b1, w2, b2, enh);

    motion_pack_k<<<1, 32, 0, stream>>>(params, Bbuf);

    conv_wmma_k<<<dim3(WDIM / 16, HDIM / 8, B), 256, 0, stream>>>(
        enh, Bbuf, intensity, (float*)d_out);
}
